// unit_mm_gcn_1864015806760
// MI455X (gfx1250) — matrix operand order repaired and checked by reading, not compiled, not run on hardware
//
#include <hip/hip_runtime.h>
#include <hip/hip_bf16.h>

// ---------------- problem constants ----------------
#define NN 32
#define CC 64
#define TT 256
#define VV 25
#define RR 8
#define OO 64
#define SSUB 3
#define TTILE 16
#define NTB (TT / TTILE)              // 16 t-blocks
#define PER_STREAM (NN * OO * TT * VV) // 13,107,200 floats

typedef __attribute__((ext_vector_type(16))) __bf16 bf16x16;
typedef __attribute__((ext_vector_type(8)))  float  f32x8;

union AF { bf16x16 v; uint4 q[2]; };

static __device__ __forceinline__ __bf16 f2bf(float x) { return (__bf16)x; }

// ---------------- workspace layout (bytes) ----------------
static const size_t WS_XM    = 0;
static const size_t WS_ABUF  = 409600;
static const size_t WS_YSUM  = 25575424;
static const size_t WS_STATS = 130433024;
static const size_t WS_BNP   = 130434048;

// ===================== kernel 0: zero the stats accumulators =====================
__global__ __launch_bounds__(256) void kinit(float* __restrict__ stats) {
    int t = threadIdx.x;
    if (t < 2 * OO * 2) stats[t] = 0.0f;
}

// ===================== kernel 1: mean over T =====================
__global__ __launch_bounds__(256) void kmean(const float* __restrict__ x1,
                                             const float* __restrict__ x2,
                                             float* __restrict__ xm) {
    int b = blockIdx.x;
    int c = b & 63, n = (b >> 6) & 31, s = b >> 11;
    const float* x = s ? x2 : x1;
    int tid = threadIdx.x;
    const float* px = x + ((size_t)((n * CC + c) * TT + tid)) * VV;

    float a[VV];
#pragma unroll
    for (int v = 0; v < VV; ++v) a[v] = px[v];
#pragma unroll
    for (int v = 0; v < VV; ++v)
        for (int off = 16; off; off >>= 1) a[v] += __shfl_down(a[v], off);

    __shared__ float red[8][VV];
    int lane = tid & 31, w = tid >> 5;
    if (lane == 0) {
#pragma unroll
        for (int v = 0; v < VV; ++v) red[w][v] = a[v];
    }
    __syncthreads();
    if (tid < VV) {
        float s0 = 0.f;
#pragma unroll
        for (int ww = 0; ww < 8; ++ww) s0 += red[ww][tid];
        xm[(size_t)b * VV + tid] = s0 * (1.0f / (float)TT);
    }
}

// ===================== kernel 2: attention matrices a1/a2 (bf16, padded) ========
__global__ __launch_bounds__(256) void kattn(const float* __restrict__ xm,
                                             const float* __restrict__ PA,
                                             const float* __restrict__ alpha,
                                             const float* __restrict__ c1w, const float* __restrict__ c1b,
                                             const float* __restrict__ c2w, const float* __restrict__ c2b,
                                             const float* __restrict__ c5w, const float* __restrict__ c5b,
                                             const float* __restrict__ c6w, const float* __restrict__ c6b,
                                             __bf16* __restrict__ abuf) {
    __shared__ float xm1l[CC * VV], xm2l[CC * VV];
    __shared__ float r1l[RR * VV], r2l[RR * VV];
    __shared__ float rell[RR * VV * VV];

    int b = blockIdx.x;
    int n = b & 31, i = b >> 5;
    int tid = threadIdx.x;

    for (int f = tid; f < CC * VV; f += 256) {
        xm1l[f] = xm[(size_t)(0 * NN + n) * (CC * VV) + f];
        xm2l[f] = xm[(size_t)(1 * NN + n) * (CC * VV) + f];
    }
    __syncthreads();

    for (int f = tid; f < RR * VV; f += 256) {
        int r = f / VV, v = f - r * VV;
        float s1 = c1b[i * RR + r], s2 = c2b[i * RR + r];
        for (int c = 0; c < CC; ++c) {
            s1 += c1w[(i * RR + r) * CC + c] * xm1l[c * VV + v];
            s2 += c2w[(i * RR + r) * CC + c] * xm2l[c * VV + v];
        }
        r1l[f] = s1;
        r2l[f] = s2;
    }
    __syncthreads();

    for (int f = tid; f < RR * VV * VV; f += 256) {
        int r = f / (VV * VV), uv = f - r * (VV * VV);
        int u = uv / VV, v = uv - u * VV;
        rell[f] = tanhf(r1l[r * VV + u] - r2l[r * VV + v]);
    }
    __syncthreads();

    float al = alpha[0];
    for (int f = tid; f < OO * 32 * 32; f += 256) {
        int o = f >> 10, u = (f >> 5) & 31, v = f & 31;
        float v1 = 0.f, v2 = 0.f;
        if (u < VV && v < VV) {
            float a1 = 0.f, a2 = 0.f;
#pragma unroll
            for (int r = 0; r < RR; ++r) {
                float rv = rell[(r * VV + u) * VV + v];
                a1 += c5w[(i * OO + o) * RR + r] * rv;
                a2 += c6w[(i * OO + o) * RR + r] * rv;
            }
            v1 = a1 + c5b[i * OO + o] + PA[(i * VV + u) * VV + v];
            v2 = a2 + c6b[i * OO + o] + al;
        }
        size_t blk = (size_t)(i * NN + n) * (OO * 32 * 32) + f;
        abuf[blk] = f2bf(v1);
        abuf[(size_t)SSUB * NN * OO * 32 * 32 + blk] = f2bf(v2);
    }
}

// ===================== kernel 3: fused conv(C->O) + graph contraction ============
// grid = N * 2 * NTB = 1024 blocks, 256 threads (8 waves); block (n, s, t-block).
// LDS: xin[(v*16+t)*72 + c] bf16 ; x3[(o*16+t)*40 + v] bf16 ; wA[o*72 + c] bf16.
__global__ __launch_bounds__(256) void kmain(const float* __restrict__ x1, const float* __restrict__ x2,
                                             const float* __restrict__ c3w, const float* __restrict__ c3b,
                                             const float* __restrict__ c4w, const float* __restrict__ c4b,
                                             const __bf16* __restrict__ abuf,
                                             float* __restrict__ ysum, float* __restrict__ stats) {
    extern __shared__ char smem[];
    __bf16* xin = (__bf16*)smem;           // 400*72 = 28800 elems
    __bf16* x3 = xin + 400 * 72;           // 1024*40 = 40960 elems
    __bf16* wA = x3 + 1024 * 40;           // 64*72  = 4608 elems

    const int b = blockIdx.x;
    const int tb = b & 15;
    const int s = (b >> 4) & 1;
    const int n = b >> 5;
    const int t0 = tb * TTILE;
    const int tid = threadIdx.x, lane = tid & 31, w = tid >> 5;
    const int half = lane >> 4, l15 = lane & 15;

    const float* x = s ? x2 : x1;
    const float* wg = s ? c4w : c3w;
    const float* bg = s ? c4b : c3b;
    const __bf16* ab = abuf + (size_t)s * SSUB * NN * OO * 32 * 32;

    // zero x3 (keeps v-padding [25,32) at zero for the K-padded z GEMM)
    for (int f = tid; f < 1024 * 40 / 2; f += 256) ((unsigned int*)x3)[f] = 0u;

    // stage the x tile into LDS as bf16, B-operand friendly layout
    for (int c = 0; c < CC; ++c) {
        const float* px = x + ((size_t)(n * CC + c) * TT + t0) * VV;
        for (int f = tid; f < TTILE * VV; f += 256) {
            int t = f / VV, v = f - t * VV;
            xin[(v * TTILE + t) * 72 + c] = f2bf(px[t * VV + v]);
        }
    }
    __syncthreads();

    f32x8 accz[8][2];
#pragma unroll
    for (int j = 0; j < 8; ++j)
#pragma unroll
        for (int ut = 0; ut < 2; ++ut)
#pragma unroll
            for (int r = 0; r < 8; ++r) accz[j][ut][r] = 0.0f;

    for (int i = 0; i < SSUB; ++i) {
        // stage conv weights (O x C) as bf16
        for (int f = tid; f < OO * CC; f += 256) {
            int o = f >> 6, c = f & 63;
            wA[o * 72 + c] = f2bf(wg[((size_t)i * OO + o) * CC + c]);
        }
        __syncthreads();

        // ---- conv stage: x3[o][t][v] = sum_c w[o][c]*xin[c][t][v] + b[o] ----
        // Software-pipelined with the first iteration peeled: the convert+store of
        // iteration v-2 sits unconditionally between the two dependent WMMAs of
        // iteration v, hiding the bf16 WMMA RAW hazard without EXEC juggling or
        // register-rotation moves.
        {
            const int otile = w & 3, obase = otile * 16, vstart = w >> 2;
            AF A0, A1;
            {
                const __bf16* wp = wA + (obase + l15) * 72;
                A0.q[0] = *(const uint4*)(wp + half * 16);
                A0.q[1] = *(const uint4*)(wp + half * 16 + 8);
                A1.q[0] = *(const uint4*)(wp + 32 + half * 16);
                A1.q[1] = *(const uint4*)(wp + 32 + half * 16 + 8);
            }
            f32x8 cb;
#pragma unroll
            for (int r = 0; r < 8; ++r) cb[r] = bg[(size_t)i * OO + obase + r + 8 * half];

            // peeled first iteration (every wave has vstart < VV)
            f32x8 dprev;
            int vprev = vstart;
            {
                AF B0, B1;
                const __bf16* bp = xin + (vstart * TTILE + l15) * 72;
                B0.q[0] = *(const uint4*)(bp + half * 16);
                B0.q[1] = *(const uint4*)(bp + half * 16 + 8);
                B1.q[0] = *(const uint4*)(bp + 32 + half * 16);
                B1.q[1] = *(const uint4*)(bp + 32 + half * 16 + 8);
                f32x8 t = __builtin_amdgcn_wmma_f32_16x16x32_bf16(false, A0.v, false, B0.v,
                                                                  (short)0, cb, false, false);
                dprev = __builtin_amdgcn_wmma_f32_16x16x32_bf16(false, A1.v, false, B1.v,
                                                                (short)0, t, false, false);
            }
            for (int v = vstart + 2; v < VV; v += 2) {
                AF B0, B1;
                const __bf16* bp = xin + (v * TTILE + l15) * 72;
                B0.q[0] = *(const uint4*)(bp + half * 16);
                B0.q[1] = *(const uint4*)(bp + half * 16 + 8);
                B1.q[0] = *(const uint4*)(bp + 32 + half * 16);
                B1.q[1] = *(const uint4*)(bp + 32 + half * 16 + 8);
                f32x8 t = __builtin_amdgcn_wmma_f32_16x16x32_bf16(false, A0.v, false, B0.v,
                                                                  (short)0, cb, false, false);
#pragma unroll
                for (int r = 0; r < 8; ++r) {
                    int o = obase + r + 8 * half;
                    x3[(o * TTILE + l15) * 40 + vprev] = f2bf(dprev[r]);
                }
                dprev = __builtin_amdgcn_wmma_f32_16x16x32_bf16(false, A1.v, false, B1.v,
                                                                (short)0, t, false, false);
                vprev = v;
            }
            // drain
#pragma unroll
            for (int r = 0; r < 8; ++r) {
                int o = obase + r + 8 * half;
                x3[(o * TTILE + l15) * 40 + vprev] = f2bf(dprev[r]);
            }
        }
        __syncthreads();

        // ---- z stage: accz[o][t][u] += sum_v x3[o][t][v]*a[o][u][v] ----
        // 16 independent accumulator chains; compiler interleaves freely.
#pragma unroll
        for (int j = 0; j < 8; ++j) {
            int o = w * 8 + j;
            AF A;
            const __bf16* ap = x3 + (o * TTILE + l15) * 40;
            A.q[0] = *(const uint4*)(ap + half * 16);
            A.q[1] = *(const uint4*)(ap + half * 16 + 8);
            const __bf16* bp = ab + (size_t)(i * NN + n) * (OO * 32 * 32) + o * (32 * 32);
#pragma unroll
            for (int ut = 0; ut < 2; ++ut) {
                AF B;
                const __bf16* bpp = bp + (ut * 16 + l15) * 32;
                B.q[0] = *(const uint4*)(bpp + half * 16);
                B.q[1] = *(const uint4*)(bpp + half * 16 + 8);
                accz[j][ut] = __builtin_amdgcn_wmma_f32_16x16x32_bf16(false, A.v, false, B.v,
                                                                      (short)0, accz[j][ut], false, false);
            }
        }
        __syncthreads();
    }

    // ---- epilogue: store pre-BN y, accumulate per-channel sum / sum-of-squares ----
#pragma unroll
    for (int j = 0; j < 8; ++j) {
        int o = w * 8 + j;
        float* yb = ysum + ((size_t)(s * NN + n) * OO + o) * (TT * VV) + (size_t)t0 * VV;
        float js1 = 0.f, js2 = 0.f;
#pragma unroll
        for (int ut = 0; ut < 2; ++ut) {
            int u = ut * 16 + l15;
            if (u < VV) {
#pragma unroll
                for (int r = 0; r < 8; ++r) {
                    int tl = r + 8 * half;
                    float val = accz[j][ut][r];
                    yb[tl * VV + u] = val;
                    js1 += val;
                    js2 += val * val;
                }
            }
        }
        for (int off = 16; off; off >>= 1) {
            js1 += __shfl_down(js1, off);
            js2 += __shfl_down(js2, off);
        }
        if (lane == 0) {
            unsafeAtomicAdd(&stats[(s * OO + o) * 2 + 0], js1);
            unsafeAtomicAdd(&stats[(s * OO + o) * 2 + 1], js2);
        }
    }
}

// ===================== kernel 4: BN parameter fold =====================
__global__ __launch_bounds__(128) void kbn(const float* __restrict__ stats,
                                           const float* __restrict__ bn1w, const float* __restrict__ bn1b,
                                           const float* __restrict__ bn2w, const float* __restrict__ bn2b,
                                           float* __restrict__ bnp) {
    int t = threadIdx.x;
    if (t >= 2 * OO) return;
    int s = t >> 6, o = t & 63;
    const float inv = 1.0f / (float)(NN * TT * VV);
    float m = stats[t * 2 + 0] * inv;
    float var = stats[t * 2 + 1] * inv - m * m;
    float wv = s ? bn2w[o] : bn1w[o];
    float bv = s ? bn2b[o] : bn1b[o];
    float sc = wv * __frsqrt_rn(var + 1e-5f);
    bnp[t * 2 + 0] = sc;
    bnp[t * 2 + 1] = bv - m * sc;
}

// ===================== kernel 5: BN + residual + ReLU (vectorized) ==============
__global__ __launch_bounds__(256) void kout(const float* __restrict__ ysum,
                                            const float* __restrict__ x1, const float* __restrict__ x2,
                                            const float* __restrict__ bnp, float* __restrict__ out) {
    size_t idx4 = (size_t)blockIdx.x * 256 + threadIdx.x;
    size_t f = idx4 * 4;
    int s = (int)(f / PER_STREAM);
    size_t rem = f - (size_t)s * PER_STREAM;
    int o = (int)((rem / (TT * VV)) & 63);
    float sc = bnp[(s * OO + o) * 2 + 0];
    float sh = bnp[(s * OO + o) * 2 + 1];
    const float4 ys = ((const float4*)ysum)[idx4];
    const float* xs = s ? x2 : x1;
    const float4 xv = *(const float4*)(xs + rem);
    float4 r;
    r.x = fmaxf(fmaf(ys.x, sc, sh) + xv.x, 0.0f);
    r.y = fmaxf(fmaf(ys.y, sc, sh) + xv.y, 0.0f);
    r.z = fmaxf(fmaf(ys.z, sc, sh) + xv.z, 0.0f);
    r.w = fmaxf(fmaf(ys.w, sc, sh) + xv.w, 0.0f);
    ((float4*)out)[idx4] = r;
}

// ===================== launcher =====================
extern "C" void kernel_launch(void* const* d_in, const int* in_sizes, int n_in,
                              void* d_out, int out_size, void* d_ws, size_t ws_size,
                              hipStream_t stream) {
    const float* x1 = (const float*)d_in[0];
    const float* x2 = (const float*)d_in[1];
    const float* PA = (const float*)d_in[2];
    const float* alpha = (const float*)d_in[3];
    const float* c1w = (const float*)d_in[4];
    const float* c1b = (const float*)d_in[5];
    const float* c2w = (const float*)d_in[6];
    const float* c2b = (const float*)d_in[7];
    const float* c3w = (const float*)d_in[8];
    const float* c3b = (const float*)d_in[9];
    const float* c4w = (const float*)d_in[10];
    const float* c4b = (const float*)d_in[11];
    const float* c5w = (const float*)d_in[12];
    const float* c5b = (const float*)d_in[13];
    const float* c6w = (const float*)d_in[14];
    const float* c6b = (const float*)d_in[15];
    const float* bn1w = (const float*)d_in[16];
    const float* bn1b = (const float*)d_in[17];
    const float* bn2w = (const float*)d_in[18];
    const float* bn2b = (const float*)d_in[19];

    char* ws = (char*)d_ws;
    float* xm = (float*)(ws + WS_XM);
    __bf16* abuf = (__bf16*)(ws + WS_ABUF);
    float* ysum = (float*)(ws + WS_YSUM);
    float* stats = (float*)(ws + WS_STATS);
    float* bnp = (float*)(ws + WS_BNP);

    kinit<<<1, 256, 0, stream>>>(stats);
    kmean<<<2 * NN * CC, 256, 0, stream>>>(x1, x2, xm);
    kattn<<<SSUB * NN, 256, 0, stream>>>(xm, PA, alpha, c1w, c1b, c2w, c2b,
                                         c5w, c5b, c6w, c6b, abuf);

    size_t ldsBytes = (size_t)(400 * 72 + 1024 * 40 + 64 * 72) * sizeof(__bf16); // 148,736 B
    (void)hipFuncSetAttribute((const void*)kmain,
                              hipFuncAttributeMaxDynamicSharedMemorySize, (int)ldsBytes);
    kmain<<<NN * 2 * NTB, 256, ldsBytes, stream>>>(x1, x2, c3w, c3b, c4w, c4b,
                                                   abuf, ysum, stats);

    kbn<<<1, 128, 0, stream>>>(stats, bn1w, bn1b, bn2w, bn2b, bnp);
    kout<<<(2 * PER_STREAM / 4) / 256, 256, 0, stream>>>(ysum, x1, x2, bnp, (float*)d_out);
}
